// ComplexMultiHeadAttention_52639119180156
// MI455X (gfx1250) — matrix operand order repaired and checked by reading, not compiled, not run on hardware
//
#include <hip/hip_runtime.h>
#include <stdint.h>

// ---------------- types ----------------
typedef __attribute__((ext_vector_type(4)))  unsigned int u32x4;
typedef __attribute__((ext_vector_type(8)))  unsigned int u32x8;
typedef __attribute__((ext_vector_type(8)))  float        v8f;
typedef __attribute__((ext_vector_type(16))) __bf16       bf16x16;

// ---------------- problem constants ----------------
#define CDIM   1024
#define CHEADS 16
#define CHD    64
#define CB     2
#define CN     2048
#define CM     (CB * CN)      // 4096 rows
#define C3D    (3 * CDIM)     // 3072
#define CSCALE 0.125f         // 64^-0.5

// ---------------- helpers ----------------
__device__ __forceinline__ v8f vzero8() {
  v8f z;
#pragma unroll
  for (int i = 0; i < 8; ++i) z[i] = 0.0f;
  return z;
}

__device__ __forceinline__ v8f wmma_bf16(u32x8 a, u32x8 b, v8f c) {
  return __builtin_amdgcn_wmma_f32_16x16x32_bf16(
      false, __builtin_bit_cast(bf16x16, a),
      false, __builtin_bit_cast(bf16x16, b),
      (short)0, c, false, false);
}

__device__ __forceinline__ uint16_t f2bf(float f) {
  uint32_t u = __builtin_bit_cast(uint32_t, f);
  u += 0x7fffu + ((u >> 16) & 1u);
  return (uint16_t)(u >> 16);
}

// A fragment (16x32 bf16, M x K), source row-major (row, k) with leading dim ld.
__device__ __forceinline__ u32x8 load_frag_a(const uint16_t* base, int row0, int k0, int ld) {
  const int lane = threadIdx.x & 31;
  const int half = lane >> 4;
  const uint16_t* p = base + (size_t)(row0 + (lane & 15)) * ld + k0 + 8 * half;
  u32x4 lo = *reinterpret_cast<const u32x4*>(p);
  u32x4 hi = *reinterpret_cast<const u32x4*>(p + 16);
  u32x8 f;
  f[0] = lo[0]; f[1] = lo[1]; f[2] = lo[2]; f[3] = lo[3];
  f[4] = hi[0]; f[5] = hi[1]; f[6] = hi[2]; f[7] = hi[3];
  return f;
}

// B fragment (32x16 bf16, K x N), source stored (N, K) row-major (k contiguous).
__device__ __forceinline__ u32x8 load_frag_b(const uint16_t* base, int n0, int k0, int ld) {
  const int lane = threadIdx.x & 31;
  const int half = lane >> 4;
  const uint16_t* p = base + (size_t)(n0 + (lane & 15)) * ld + k0 + 8 * half;
  u32x4 lo = *reinterpret_cast<const u32x4*>(p);
  u32x4 hi = *reinterpret_cast<const u32x4*>(p + 16);
  u32x8 f;
  f[0] = lo[0]; f[1] = lo[1]; f[2] = lo[2]; f[3] = lo[3];
  f[4] = hi[0]; f[5] = hi[1]; f[6] = hi[2]; f[7] = hi[3];
  return f;
}

// A fragment from LDS tile (16 rows x 32 k, bf16, row stride ld halfwords)
__device__ __forceinline__ u32x8 load_frag_a_lds(const uint16_t* base, int ld) {
  const int lane = threadIdx.x & 31;
  const int half = lane >> 4;
  const uint16_t* p = base + (lane & 15) * ld + 8 * half;
  u32x8 f;
#pragma unroll
  for (int j = 0; j < 4; ++j) f[j]     = *reinterpret_cast<const uint32_t*>(p + 2 * j);
#pragma unroll
  for (int j = 0; j < 4; ++j) f[4 + j] = *reinterpret_cast<const uint32_t*>(p + 16 + 2 * j);
  return f;
}

// ---------------- kernels ----------------
__global__ __launch_bounds__(256) void cvt_f32_bf16(const float* __restrict__ in,
                                                    uint16_t* __restrict__ out, int n) {
  int i = blockIdx.x * blockDim.x + threadIdx.x;
  if (i < n) out[i] = f2bf(in[i]);
}

// C = A(M x K) * W(N x K)^T + bias.  One wave computes a 32x64 tile.
// MODE 0: scatter qkv columns into q (B,H,N,hd), k (B,H,N,hd), vT (B,H,hd,N), bf16.
// MODE 1: dense fp32 output (M x Ncols).
template <int MODE>
__global__ __launch_bounds__(32) void gemm32x64(
    const uint16_t* __restrict__ A, const uint16_t* __restrict__ W,
    const float* __restrict__ bias, float* __restrict__ outF,
    uint16_t* __restrict__ qb, uint16_t* __restrict__ kb,
    uint16_t* __restrict__ vtb, int Ncols, int K) {
  const int lane  = threadIdx.x & 31;
  const int half  = lane >> 4;
  const int nBase = blockIdx.x * 64;
  const int mBase = blockIdx.y * 32;

  v8f acc[2][4];
#pragma unroll
  for (int i = 0; i < 2; ++i)
#pragma unroll
    for (int t = 0; t < 4; ++t) acc[i][t] = vzero8();

  for (int k0 = 0; k0 < K; k0 += 32) {
    u32x8 a0 = load_frag_a(A, mBase, k0, K);
    u32x8 a1 = load_frag_a(A, mBase + 16, k0, K);
#pragma unroll
    for (int t = 0; t < 4; ++t) {
      u32x8 b = load_frag_b(W, nBase + 16 * t, k0, K);
      acc[0][t] = wmma_bf16(a0, b, acc[0][t]);
      acc[1][t] = wmma_bf16(a1, b, acc[1][t]);
    }
  }

#pragma unroll
  for (int i = 0; i < 2; ++i)
#pragma unroll
    for (int t = 0; t < 4; ++t) {
      const int n = nBase + 16 * t + (lane & 15);
      const float bv = bias[n];
#pragma unroll
      for (int r = 0; r < 8; ++r) {
        const int m = mBase + 16 * i + r + 8 * half;
        const float v = acc[i][t][r] + bv;
        if (MODE == 0) {
          const int sel = n >> 10, h = (n >> 6) & 15, d = n & 63;
          const int b = m >> 11, tok = m & 2047;
          const uint16_t bf = f2bf(v);
          if (sel == 0)
            qb[((size_t)(b * CHEADS + h) * CN + tok) * CHD + d] = bf;
          else if (sel == 1)
            kb[((size_t)(b * CHEADS + h) * CN + tok) * CHD + d] = bf;
          else
            vtb[((size_t)(b * CHEADS + h) * CHD + d) * CN + tok] = bf;
        } else {
          outF[(size_t)m * Ncols + n] = v;
        }
      }
    }
}

// Fused complex flash attention: one wave = 16 query rows of one (b,h).
__global__ __launch_bounds__(32) void attn_kernel(
    const uint16_t* __restrict__ qr, const uint16_t* __restrict__ kr,
    const uint16_t* __restrict__ vtr,
    const uint16_t* __restrict__ qi, const uint16_t* __restrict__ ki,
    const uint16_t* __restrict__ vti,
    uint16_t* __restrict__ aor, uint16_t* __restrict__ aoi) {
  __shared__ uint16_t pls[16 * 40];  // P tile, 16 x 32 bf16, stride 40 halfwords

  const int lane  = threadIdx.x & 31;
  const int half  = lane >> 4;
  const int bh    = blockIdx.y;
  const int qBase = blockIdx.x * 16;

  const uint16_t* qrp = qr + (size_t)bh * CN * CHD;
  const uint16_t* krp = kr + (size_t)bh * CN * CHD;
  const uint16_t* qip = qi + (size_t)bh * CN * CHD;
  const uint16_t* kip = ki + (size_t)bh * CN * CHD;
  const uint16_t* vtrp = vtr + (size_t)bh * CHD * CN;
  const uint16_t* vtip = vti + (size_t)bh * CHD * CN;

  u32x8 aQr[2], aQi[2], aQrN[2];
#pragma unroll
  for (int kc = 0; kc < 2; ++kc) {
    aQr[kc] = load_frag_a(qrp, qBase, kc * 32, CHD);
    aQi[kc] = load_frag_a(qip, qBase, kc * 32, CHD);
    aQrN[kc] = aQr[kc] ^ 0x80008000u;  // negate packed bf16 pair
  }

  v8f oR[4], oI[4];
#pragma unroll
  for (int t = 0; t < 4; ++t) { oR[t] = vzero8(); oI[t] = vzero8(); }
  float rm[8], rl[8];
#pragma unroll
  for (int r = 0; r < 8; ++r) { rm[r] = -1e30f; rl[r] = 0.0f; }

  for (int kb0 = 0; kb0 < CN; kb0 += 32) {
    v8f mag[2];
#pragma unroll
    for (int sub = 0; sub < 2; ++sub) {
      const int kk = kb0 + sub * 16;
      v8f sr = vzero8(), si = vzero8();
#pragma unroll
      for (int kc = 0; kc < 2; ++kc) {
        u32x8 bKr = load_frag_b(krp, kk, kc * 32, CHD);
        u32x8 bKi = load_frag_b(kip, kk, kc * 32, CHD);
        sr = wmma_bf16(aQr[kc], bKr, sr);
        sr = wmma_bf16(aQi[kc], bKi, sr);
        si = wmma_bf16(aQi[kc], bKr, si);
        si = wmma_bf16(aQrN[kc], bKi, si);
      }
#pragma unroll
      for (int r = 0; r < 8; ++r)
        mag[sub][r] = CSCALE * sqrtf(sr[r] * sr[r] + si[r] * si[r]);
    }

    // online softmax state update (row reductions within 16-lane halves)
    float nm[8], al[8];
#pragma unroll
    for (int r = 0; r < 8; ++r) {
      float v = fmaxf(mag[0][r], mag[1][r]);
      v = fmaxf(v, __shfl_xor(v, 1, 16));
      v = fmaxf(v, __shfl_xor(v, 2, 16));
      v = fmaxf(v, __shfl_xor(v, 4, 16));
      v = fmaxf(v, __shfl_xor(v, 8, 16));
      nm[r] = fmaxf(rm[r], v);
      al[r] = __expf(rm[r] - nm[r]);
      rm[r] = nm[r];
    }

    // P = exp(mag - m_new): keep in regs for rowsum, stage to LDS for refragment
#pragma unroll
    for (int sub = 0; sub < 2; ++sub)
#pragma unroll
      for (int r = 0; r < 8; ++r) {
        const float p = __expf(mag[sub][r] - nm[r]);
        mag[sub][r] = p;
        pls[(r + 8 * half) * 40 + (lane & 15) + 16 * sub] = f2bf(p);
      }

#pragma unroll
    for (int r = 0; r < 8; ++r) {
      float s = mag[0][r] + mag[1][r];
      s += __shfl_xor(s, 1, 16);
      s += __shfl_xor(s, 2, 16);
      s += __shfl_xor(s, 4, 16);
      s += __shfl_xor(s, 8, 16);
      rl[r] = rl[r] * al[r] + s;
    }

#pragma unroll
    for (int t = 0; t < 4; ++t)
#pragma unroll
      for (int r = 0; r < 8; ++r) { oR[t][r] *= al[r]; oI[t][r] *= al[r]; }

    asm volatile("s_wait_dscnt 0" ::: "memory");
    u32x8 ap = load_frag_a_lds(pls, 40);

#pragma unroll
    for (int t = 0; t < 4; ++t) {
      u32x8 bVr = load_frag_b(vtrp, t * 16, kb0, CN);
      oR[t] = wmma_bf16(ap, bVr, oR[t]);
      u32x8 bVi = load_frag_b(vtip, t * 16, kb0, CN);
      oI[t] = wmma_bf16(ap, bVi, oI[t]);
    }
    asm volatile("" ::: "memory");
  }

  // epilogue: normalize and write (B*N, D) bf16 with column h*64+d
  const int b = bh >> 4, h = bh & 15;
#pragma unroll
  for (int t = 0; t < 4; ++t) {
    const int col = h * CHD + t * 16 + (lane & 15);
#pragma unroll
    for (int r = 0; r < 8; ++r) {
      const int row = b * CN + qBase + r + 8 * half;
      const float inv = 1.0f / rl[r];
      aor[(size_t)row * CDIM + col] = f2bf(oR[t][r] * inv);
      aoi[(size_t)row * CDIM + col] = f2bf(oI[t][r] * inv);
    }
  }
}

// ---------------- host launch ----------------
extern "C" void kernel_launch(void* const* d_in, const int* in_sizes, int n_in,
                              void* d_out, int out_size, void* d_ws, size_t ws_size,
                              hipStream_t stream) {
  const float* x_r    = (const float*)d_in[0];
  const float* x_i    = (const float*)d_in[1];
  const float* Wqkv_r = (const float*)d_in[2];
  const float* bqkv_r = (const float*)d_in[3];
  const float* Wqkv_i = (const float*)d_in[4];
  const float* bqkv_i = (const float*)d_in[5];
  const float* Wout_r = (const float*)d_in[6];
  const float* bout_r = (const float*)d_in[7];
  const float* Wout_i = (const float*)d_in[8];
  const float* bout_i = (const float*)d_in[9];

  char* ws = (char*)d_ws;
  size_t off = 0;
  auto take = [&](size_t bytes) -> char* {
    char* p = ws + off;
    off += (bytes + 255) & ~(size_t)255;
    return p;
  };

  const size_t NX   = (size_t)CM * CDIM;               // 4,194,304
  const size_t NWQ  = (size_t)C3D * CDIM;              // 3,145,728
  const size_t NWO  = (size_t)CDIM * CDIM;             // 1,048,576
  const size_t NQKV = (size_t)CB * CHEADS * CN * CHD;  // 4,194,304

  uint16_t* XRB = (uint16_t*)take(NX * 2);
  uint16_t* XIB = (uint16_t*)take(NX * 2);
  uint16_t* WQR = (uint16_t*)take(NWQ * 2);
  uint16_t* WQI = (uint16_t*)take(NWQ * 2);
  uint16_t* WOR = (uint16_t*)take(NWO * 2);
  uint16_t* WOI = (uint16_t*)take(NWO * 2);
  uint16_t* QR  = (uint16_t*)take(NQKV * 2);
  uint16_t* KR  = (uint16_t*)take(NQKV * 2);
  uint16_t* VTR = (uint16_t*)take(NQKV * 2);
  uint16_t* QI  = (uint16_t*)take(NQKV * 2);
  uint16_t* KI  = (uint16_t*)take(NQKV * 2);
  uint16_t* VTI = (uint16_t*)take(NQKV * 2);
  uint16_t* AOR = (uint16_t*)take(NX * 2);
  uint16_t* AOI = (uint16_t*)take(NX * 2);

  // stage 0: fp32 -> bf16
  cvt_f32_bf16<<<(int)(NX / 256), 256, 0, stream>>>(x_r, XRB, (int)NX);
  cvt_f32_bf16<<<(int)(NX / 256), 256, 0, stream>>>(x_i, XIB, (int)NX);
  cvt_f32_bf16<<<(int)(NWQ / 256), 256, 0, stream>>>(Wqkv_r, WQR, (int)NWQ);
  cvt_f32_bf16<<<(int)(NWQ / 256), 256, 0, stream>>>(Wqkv_i, WQI, (int)NWQ);
  cvt_f32_bf16<<<(int)(NWO / 256), 256, 0, stream>>>(Wout_r, WOR, (int)NWO);
  cvt_f32_bf16<<<(int)(NWO / 256), 256, 0, stream>>>(Wout_i, WOI, (int)NWO);

  // stage 1: QKV projections (scatter into attention layouts)
  dim3 gq(C3D / 64, CM / 32);
  gemm32x64<0><<<gq, 32, 0, stream>>>(XRB, WQR, bqkv_r, nullptr, QR, KR, VTR, C3D, CDIM);
  gemm32x64<0><<<gq, 32, 0, stream>>>(XIB, WQI, bqkv_i, nullptr, QI, KI, VTI, C3D, CDIM);

  // stage 2: fused complex flash attention
  attn_kernel<<<dim3(CN / 16, CB * CHEADS), 32, 0, stream>>>(QR, KR, VTR, QI, KI, VTI, AOR, AOI);

  // stage 3: output projections -> fp32 d_out (real half then imag half)
  float* out_r = (float*)d_out;
  float* out_i = out_r + NX;
  dim3 go(CDIM / 64, CM / 32);
  gemm32x64<1><<<go, 32, 0, stream>>>(AOR, WOR, bout_r, out_r, nullptr, nullptr, nullptr, CDIM, CDIM);
  gemm32x64<1><<<go, 32, 0, stream>>>(AOI, WOI, bout_i, out_i, nullptr, nullptr, nullptr, CDIM, CDIM);
}
